// Comb1_18425409699929
// MI455X (gfx1250) — matrix operand order repaired and checked by reading, not compiled, not run on hardware
//
#include <hip/hip_runtime.h>

// ---------------- problem constants ----------------
#define BSZ   8
#define NDIM  200           // T*N nodes
#define NP    224           // padded nodes: 14*16, 7*32
#define CDIM  256           // channels: 16*16, 8*32
#define EPSV  1e-12f
#define LNEPS 1e-5f
#define SLOPE 0.01f
#define SCALE 5.0f

typedef __bf16 bf16;
typedef bf16  v16bf  __attribute__((ext_vector_type(16)));
typedef bf16  bf16x8 __attribute__((ext_vector_type(8)));
typedef float v8f    __attribute__((ext_vector_type(8)));

// ---------------------------------------------------------------------------
// NT bf16 WMMA GEMM: C[M,N] = A[M,K] @ Bsrc^T, Bsrc row-major [N,K].
// Fragment layouts (ISA 7.12.2, 16-bit A 16x32 / B 32x16):
//   A lane group (lane&16): K offsets {khalfA+0..7, khalfA+16..23} -> two b128
//   B lane col = lane&15:   K offsets {khalfB+0..7, khalfB+16..23} -> two b128
// One wave computes TWO 16x16 N-tiles sharing the A fragment; 2 waves/block
// along M. K is compile-time -> fully unrolled WMMA chain.
// Outputs (compile-time selected): f32 row-major, bf16 row-major,
// bf16 transposed (for feeding the next NT GEMM as Bsrc).
// ---------------------------------------------------------------------------
template <int KD, bool WF, bool WB, bool WT>
__global__ void wmma_gemm_nt(const bf16* __restrict__ Aall, long strideA, int lda,
                             const bf16* __restrict__ Ball, long strideB, int ldb,
                             float* __restrict__ outF, bf16* __restrict__ outB,
                             bf16* __restrict__ outT,
                             long strideO, int ldc, long strideT, int ldt)
{
    const int lane   = threadIdx.x;                 // 0..31 (wave32)
    const int tileM  = blockIdx.y * 2 + threadIdx.y;
    const int tileN0 = blockIdx.x * 2;              // this wave: tileN0, tileN0+1
    const int b      = blockIdx.z;

    const bf16* A  = Aall + (size_t)b * strideA;
    const bf16* Bm = Ball + (size_t)b * strideB;

    const int rowA   = lane & 15;
    const int khalfA = (lane >> 4) * 8;
    const int colB   = lane & 15;
    const int khalfB = (lane >> 4) * 8;

    const bf16* arow  = A  + (size_t)(tileM * 16 + rowA) * lda;
    const bf16* brow0 = Bm + (size_t)(tileN0 * 16 + colB) * ldb;
    const bf16* brow1 = brow0 + (size_t)16 * ldb;

    v8f acc0 = {}, acc1 = {};
#pragma unroll
    for (int kk = 0; kk < KD; kk += 32) {
        const bf16x8 alo = *reinterpret_cast<const bf16x8*>(arow + kk + khalfA);
        const bf16x8 ahi = *reinterpret_cast<const bf16x8*>(arow + kk + khalfA + 16);
        const v16bf  af  = __builtin_shufflevector(alo, ahi,
                              0,1,2,3,4,5,6,7,8,9,10,11,12,13,14,15);

        const bf16x8 b0l = *reinterpret_cast<const bf16x8*>(brow0 + kk + khalfB);
        const bf16x8 b0h = *reinterpret_cast<const bf16x8*>(brow0 + kk + khalfB + 16);
        const v16bf  bf0 = __builtin_shufflevector(b0l, b0h,
                              0,1,2,3,4,5,6,7,8,9,10,11,12,13,14,15);

        const bf16x8 b1l = *reinterpret_cast<const bf16x8*>(brow1 + kk + khalfB);
        const bf16x8 b1h = *reinterpret_cast<const bf16x8*>(brow1 + kk + khalfB + 16);
        const v16bf  bf1 = __builtin_shufflevector(b1l, b1h,
                              0,1,2,3,4,5,6,7,8,9,10,11,12,13,14,15);

        acc0 = __builtin_amdgcn_wmma_f32_16x16x32_bf16(
                   false, af, false, bf0, (short)0, acc0, false, false);
        acc1 = __builtin_amdgcn_wmma_f32_16x16x32_bf16(
                   false, af, false, bf1, (short)0, acc1, false, false);
    }

    // C/D layout: VGPR v -> row M = v + 8*(lane>=16), col N = lane&15
    const int colb  = lane & 15;
    const int rbase = tileM * 16 + ((lane >> 4) ? 8 : 0);
    const int col0  = tileN0 * 16 + colb;
    const int col1  = col0 + 16;
#pragma unroll
    for (int v = 0; v < 8; ++v) {
        const int row = rbase + v;
        if (WF) {
            float* o = outF + (size_t)b * strideO + (size_t)row * ldc;
            o[col0] = acc0[v];
            o[col1] = acc1[v];
        }
        if (WB) {
            bf16* o = outB + (size_t)b * strideO + (size_t)row * ldc;
            o[col0] = (bf16)acc0[v];
            o[col1] = (bf16)acc1[v];
        }
        if (WT) {
            bf16* o = outT + (size_t)b * strideT;
            o[(size_t)col0 * ldt + row] = (bf16)acc0[v];
            o[(size_t)col1 * ldt + row] = (bf16)acc1[v];
        }
    }
}

// ---------------------------------------------------------------------------
// L2-normalize rows of lf[B,NDIM,CDIM] -> bf16 padded [B,NP,CDIM] (+ raw bf16
// copy and optional f32 normalized copy). Pad rows are zeroed.
// ---------------------------------------------------------------------------
__global__ void l2norm_pad_kernel(const float* __restrict__ lf,
                                  bf16* __restrict__ nf_bf,
                                  bf16* __restrict__ lf_bf,
                                  float* __restrict__ nf_f32)
{
    const int i = blockIdx.x;      // 0..NP-1
    const int b = blockIdx.y;
    const int c = threadIdx.x;     // 0..255
    __shared__ float red[256];

    const size_t po = ((size_t)b * NP + i) * CDIM + c;
    if (i >= NDIM) {               // uniform per block -> no divergent barrier
        nf_bf[po] = (bf16)0.0f;
        if (lf_bf) lf_bf[po] = (bf16)0.0f;
        return;
    }
    const float x = lf[((size_t)b * NDIM + i) * CDIM + c];
    red[c] = x * x;
    __syncthreads();
    for (int s = 128; s > 0; s >>= 1) { if (c < s) red[c] += red[c + s]; __syncthreads(); }
    const float inv = 1.0f / fmaxf(sqrtf(red[0]), EPSV);
    const float nx  = x * inv;
    nf_bf[po] = (bf16)nx;
    if (lf_bf)  lf_bf[po] = (bf16)x;
    if (nf_f32) nf_f32[((size_t)b * NDIM + i) * CDIM + c] = nx;
}

// ---------------------------------------------------------------------------
// Row-wise attention from raw logits L[B,NP,NP]:
//  mode 0 (diff_prop): e=exp(5(L-max)); e[i,i]=0; A=e/max(sum,EPS); s=sum/max(sum,EPS)
//  mode 1 (feat_aggr): softmax(5*L) (shift-invariant); s=1
// ---------------------------------------------------------------------------
__global__ void attn_kernel(const float* __restrict__ L, bf16* __restrict__ Abf,
                            float* __restrict__ s, int mode)
{
    const int i = blockIdx.x;
    const int b = blockIdx.y;
    const int t = threadIdx.x;
    __shared__ float red[256];
    const size_t rb = ((size_t)b * NP + i) * NP;

    if (i >= NDIM) {
        for (int j = t; j < NP; j += 256) Abf[rb + j] = (bf16)0.0f;
        if (t == 0) s[(size_t)b * NP + i] = 0.0f;
        return;
    }
    float m = -3.0e38f;
    for (int j = t; j < NDIM; j += 256) m = fmaxf(m, L[rb + j]);
    red[t] = m; __syncthreads();
    for (int q = 128; q > 0; q >>= 1) { if (t < q) red[t] = fmaxf(red[t], red[t + q]); __syncthreads(); }
    m = red[0];
    __syncthreads();

    float sum = 0.0f;
    for (int j = t; j < NDIM; j += 256) {
        float e = expf(SCALE * (L[rb + j] - m));
        if (mode == 0 && j == i) e = 0.0f;
        sum += e;
    }
    red[t] = sum; __syncthreads();
    for (int q = 128; q > 0; q >>= 1) { if (t < q) red[t] += red[t + q]; __syncthreads(); }
    sum = red[0];

    const float denom = (mode == 0) ? fmaxf(sum, EPSV) : sum;
    const float inv   = 1.0f / denom;
    for (int j = t; j < NP; j += 256) {
        float a = 0.0f;
        if (j < NDIM) {
            float e = expf(SCALE * (L[rb + j] - m));
            if (mode == 0 && j == i) e = 0.0f;
            a = e * inv;
        }
        Abf[rb + j] = (bf16)a;
    }
    if (t == 0) s[(size_t)b * NP + i] = (mode == 0) ? sum * inv : 1.0f;
}

// ---------------------------------------------------------------------------
// Epilogue: combine + LayerNorm + LeakyReLU.
//  mode 0: x = s_i*(Q + aff_b) - AQ   (collapsed pairwise-difference form)
//  mode 1: x = AQ + aff_b             (softmax rows sum to 1)
// ---------------------------------------------------------------------------
__global__ void epilogue_kernel(const float* __restrict__ Qf, const float* __restrict__ AQf,
                                const float* __restrict__ s,  const float* __restrict__ affb,
                                const float* __restrict__ lng, const float* __restrict__ lnb,
                                float* __restrict__ out, int mode)
{
    const int i = blockIdx.x;   // < NDIM
    const int b = blockIdx.y;
    const int c = threadIdx.x;
    __shared__ float red[256];
    const size_t rp = ((size_t)b * NP + i) * CDIM + c;

    float x;
    if (mode == 0) {
        const float si = s[(size_t)b * NP + i];
        x = si * (Qf[rp] + affb[c]) - AQf[rp];
    } else {
        x = AQf[rp] + affb[c];
    }
    red[c] = x; __syncthreads();
    for (int q = 128; q > 0; q >>= 1) { if (c < q) red[c] += red[c + q]; __syncthreads(); }
    const float mean = red[0] * (1.0f / CDIM);
    __syncthreads();
    const float d0 = x - mean;
    red[c] = d0 * d0; __syncthreads();
    for (int q = 128; q > 0; q >>= 1) { if (c < q) red[c] += red[c + q]; __syncthreads(); }
    const float var = red[0] * (1.0f / CDIM);

    float y = d0 * rsqrtf(var + LNEPS) * lng[c] + lnb[c];
    y = (y >= 0.0f) ? y : SLOPE * y;
    out[((size_t)b * NDIM + i) * CDIM + c] = y;
}

// ---------------------------------------------------------------------------
// node_att, collapsed: att_i = sigmoid(nf_i . u_b + p_i . w_pos + b),
// u_b[c] = sum_j w_j * nf[b,j,c]
// ---------------------------------------------------------------------------
__global__ void natt_u_kernel(const float* __restrict__ nf, const float* __restrict__ naw,
                              float* __restrict__ u)
{
    const int idx = blockIdx.x * 256 + threadIdx.x;   // b*CDIM + c
    const int b = idx / CDIM, c = idx % CDIM;
    if (b >= BSZ) return;
    float acc = 0.0f;
    for (int j = 0; j < NDIM; ++j)
        acc += naw[j] * nf[((size_t)b * NDIM + j) * CDIM + c];
    u[idx] = acc;
}

__global__ void natt_apply_kernel(const float* __restrict__ lf, const float* __restrict__ nf,
                                  const float* __restrict__ u,  const float* __restrict__ pos,
                                  const float* __restrict__ naw, const float* __restrict__ nab,
                                  float* __restrict__ out)
{
    const int i = blockIdx.x, b = blockIdx.y, c = threadIdx.x;
    __shared__ float red[256];
    __shared__ float attsh;
    const size_t off = ((size_t)b * NDIM + i) * CDIM;

    red[c] = nf[off + c] * u[(size_t)b * CDIM + c];
    __syncthreads();
    for (int q = 128; q > 0; q >>= 1) { if (c < q) red[c] += red[c + q]; __syncthreads(); }
    if (c == 0) {
        float logit = red[0];
        const float* p = pos + ((size_t)b * NDIM + i) * 9;
        for (int k = 0; k < 9; ++k) logit += p[k] * naw[NDIM + k];
        logit += nab[0];
        attsh = 1.0f / (1.0f + expf(-logit));
    }
    __syncthreads();
    out[off + c] = lf[off + c] * attsh;
}

__global__ void f32_to_bf16_kernel(const float* __restrict__ in, bf16* __restrict__ out, int n)
{
    const int idx = blockIdx.x * 256 + threadIdx.x;
    if (idx < n) out[idx] = (bf16)in[idx];
}

// out[d*CDIM + c] = in[c*CDIM + d]  (once-per-launch 256x256 weight transpose)
__global__ void transpose_f32_to_bf16_kernel(const float* __restrict__ in,
                                             bf16* __restrict__ out)
{
    const int idx = blockIdx.x * 256 + threadIdx.x;
    const int d = idx / CDIM, c = idx % CDIM;
    out[idx] = (bf16)in[(size_t)c * CDIM + d];
}

// ---------------------------------------------------------------------------
extern "C" void kernel_launch(void* const* d_in, const int* in_sizes, int n_in,
                              void* d_out, int out_size, void* d_ws, size_t ws_size,
                              hipStream_t stream)
{
    const float* local_feat = (const float*)d_in[0];
    const float* pos        = (const float*)d_in[2];
    const float* dp_adj  = (const float*)d_in[3];
    const float* dp_aff  = (const float*)d_in[4];
    const float* dp_affb = (const float*)d_in[5];
    const float* dp_lng  = (const float*)d_in[6];
    const float* dp_lnb  = (const float*)d_in[7];
    const float* fa_adj  = (const float*)d_in[8];
    const float* fa_aff  = (const float*)d_in[9];
    const float* fa_affb = (const float*)d_in[10];
    const float* fa_lng  = (const float*)d_in[11];
    const float* fa_lnb  = (const float*)d_in[12];
    const float* na_w    = (const float*)d_in[13];
    const float* na_b    = (const float*)d_in[14];

    // carve workspace
    char* p = (char*)d_ws;
    auto alloc = [&](size_t bytes) -> void* {
        void* r = (void*)p;
        p += (bytes + 255) & ~(size_t)255;
        return r;
    };
    bf16*  nf_bf   = (bf16*) alloc((size_t)BSZ * NP * CDIM * 2);
    bf16*  lf_bf   = (bf16*) alloc((size_t)BSZ * NP * CDIM * 2);
    bf16*  m1_bf   = (bf16*) alloc((size_t)BSZ * NP * CDIM * 2);
    bf16*  qT_bf   = (bf16*) alloc((size_t)BSZ * CDIM * NP * 2);   // Q^T per batch
    bf16*  a_bf    = (bf16*) alloc((size_t)BSZ * NP * NP * 2);
    float* Lf      = (float*)alloc((size_t)BSZ * NP * NP * 4);
    float* Qf      = (float*)alloc((size_t)BSZ * NP * CDIM * 4);
    float* AQf     = (float*)alloc((size_t)BSZ * NP * CDIM * 4);
    float* sbuf    = (float*)alloc((size_t)BSZ * NP * 4);
    float* lf0     = (float*)alloc((size_t)BSZ * NDIM * CDIM * 4);
    float* lf1     = (float*)alloc((size_t)BSZ * NDIM * CDIM * 4);
    float* nf32    = (float*)alloc((size_t)BSZ * NDIM * CDIM * 4);
    float* ubuf    = (float*)alloc((size_t)BSZ * CDIM * 4);
    bf16*  w_adjT0 = (bf16*) alloc((size_t)CDIM * CDIM * 2);       // adj^T (NT form)
    bf16*  w_aff0  = (bf16*) alloc((size_t)CDIM * CDIM * 2);       // aff as-is (already ^T use)
    bf16*  w_adjT1 = (bf16*) alloc((size_t)CDIM * CDIM * 2);
    bf16*  w_aff1  = (bf16*) alloc((size_t)CDIM * CDIM * 2);

    const int we = CDIM * CDIM;
    transpose_f32_to_bf16_kernel<<<we / 256, 256, 0, stream>>>(dp_adj, w_adjT0);
    f32_to_bf16_kernel        <<<we / 256, 256, 0, stream>>>(dp_aff, w_aff0, we);
    transpose_f32_to_bf16_kernel<<<we / 256, 256, 0, stream>>>(fa_adj, w_adjT1);
    f32_to_bf16_kernel        <<<we / 256, 256, 0, stream>>>(fa_aff, w_aff1, we);

    const long sNC = (long)NP * CDIM;    // per-batch stride, feature mats
    const long sNN = (long)NP * NP;      // per-batch stride, attention mats
    const long sCN = (long)CDIM * NP;    // per-batch stride, Q^T

    const dim3 blk(32, 2);               // 2 waves along M
    const dim3 gFeat(CDIM / 32, NP / 32, BSZ);   // N=CDIM: 8 x 7
    const dim3 gAttn(NP   / 32, NP / 32, BSZ);   // N=NP:   7 x 7

    auto run_module = [&](const float* lf_in, float* lf_out,
                          const bf16* adjT_bf, const bf16* aff_bf,
                          const float* affb, const float* lng, const float* lnb, int mode)
    {
        l2norm_pad_kernel<<<dim3(NP, BSZ), 256, 0, stream>>>(lf_in, nf_bf, lf_bf, nullptr);
        // M1 = nf @ adj  == nf @ (adj^T)^T            [NP,CDIM], K=256  -> bf16
        wmma_gemm_nt<256, false, true, false><<<gFeat, blk, 0, stream>>>(
            nf_bf, sNC, CDIM, adjT_bf, 0L, CDIM,
            nullptr, m1_bf, nullptr, sNC, CDIM, 0L, 0);
        // L = M1 @ nf^T                               [NP,NP],   K=256  -> f32
        wmma_gemm_nt<256, true, false, false><<<gAttn, blk, 0, stream>>>(
            m1_bf, sNC, CDIM, nf_bf, sNC, CDIM,
            Lf, nullptr, nullptr, sNN, NP, 0L, 0);
        attn_kernel<<<dim3(NP, BSZ), 256, 0, stream>>>(Lf, a_bf, sbuf, mode);
        // Q = lf @ aff^T                              [NP,CDIM], K=256  -> f32 + bf16^T
        wmma_gemm_nt<256, true, false, true><<<gFeat, blk, 0, stream>>>(
            lf_bf, sNC, CDIM, aff_bf, 0L, CDIM,
            Qf, nullptr, qT_bf, sNC, CDIM, sCN, NP);
        // AQ = A @ Q == A @ (Q^T)^T                   [NP,CDIM], K=224  -> f32
        wmma_gemm_nt<224, true, false, false><<<gFeat, blk, 0, stream>>>(
            a_bf, sNN, NP, qT_bf, sCN, NP,
            AQf, nullptr, nullptr, sNC, CDIM, 0L, 0);
        epilogue_kernel<<<dim3(NDIM, BSZ), 256, 0, stream>>>(
            Qf, AQf, sbuf, affb, lng, lnb, lf_out, mode);
    };

    run_module(local_feat, lf0, w_adjT0, w_aff0, dp_affb, dp_lng, dp_lnb, 0);
    run_module(lf0,        lf1, w_adjT1, w_aff1, fa_affb, fa_lng, fa_lnb, 1);

    // node_att on feat_aggr output
    l2norm_pad_kernel<<<dim3(NP, BSZ), 256, 0, stream>>>(lf1, nf_bf, nullptr, nf32);
    natt_u_kernel<<<(BSZ * CDIM) / 256, 256, 0, stream>>>(nf32, na_w, ubuf);
    natt_apply_kernel<<<dim3(NDIM, BSZ), 256, 0, stream>>>(
        lf1, nf32, ubuf, pos, na_w, na_b, (float*)d_out);
}
